// BPNKernel_28028956574294
// MI455X (gfx1250) — matrix operand order repaired and checked by reading, not compiled, not run on hardware
//
#include <hip/hip_runtime.h>
#include <math.h>

typedef _Float16 h16;
typedef __attribute__((ext_vector_type(2)))  _Float16 h16x2;
typedef __attribute__((ext_vector_type(16))) _Float16 v16h;
typedef __attribute__((ext_vector_type(8)))  _Float16 v8h;
typedef __attribute__((ext_vector_type(8)))  float    v8f;

// ---------------------------------------------------------------------------
// f32 -> f16 convert with channel zero-padding: y has ntotal elements,
// first nsrc are converted from x, the rest written as 0 (pad channels).
// ---------------------------------------------------------------------------
__global__ void cvt_f32_f16_k(const float* __restrict__ x, h16* __restrict__ y,
                              int nsrc, int ntotal) {
    int i = blockIdx.x * 256 + threadIdx.x;
    if (i >= ntotal) return;
    y[i] = (i < nsrc) ? (h16)x[i] : (h16)0.0f;
}

// ---------------------------------------------------------------------------
// Repack conv weight [co][ci][3][3] fp32 -> f16 [tap][cop][Cinp]:
//   out[(tap*Coutp + co)*Cinp + ci] = w[((co*Cin + ci)*9) + tap]
// Zero pad ci -> Cinp (mult of 32) and co -> Coutp (mult of 64) so the WMMA
// A-slabs are aligned b128 loads with no runtime masking.
// ---------------------------------------------------------------------------
__global__ void wpack_k(const float* __restrict__ w, h16* __restrict__ out,
                        int Cin, int Cout, int Cinp, int Coutp, int total) {
    int i = blockIdx.x * 256 + threadIdx.x;
    if (i >= total) return;
    int per_tap = Coutp * Cinp;
    int tap = i / per_tap;
    int rem = i - tap * per_tap;
    int co  = rem / Cinp;
    int ci  = rem - co * Cinp;
    float v = 0.0f;
    if (co < Cout && ci < Cin)
        v = w[((size_t)co * Cin + ci) * 9 + tap];
    out[i] = (h16)v;
}

// ---------------------------------------------------------------------------
// Tap-decomposed implicit-GEMM 3x3 SAME conv via v_wmma_f32_16x16x32_f16.
// Block = 128 threads = 4 wave32.
//   M tile: 64 out-channels (wave w -> [blockIdx.y*64 + 16w, +16))
//   N tile: 64 pixels      (each wave: 4 WMMAs over 16-pixel groups)
//   K loop: 9 taps x (Cinp/32) chunks, flattened + LDS double-buffered:
//           one barrier per chunk, staging overlapped with WMMA.
// Activation buffers are guaranteed to hold Cinp (mult-of-32) channels with
// pad channels zeroed, so every staged load is unconditionally in-bounds:
// no branches, no div/mod, pointer-strided gathers.
// mode 0: out = f16 relu(conv+bias). mode 1: out = f32 tanh(relu(conv+bias)).
// ---------------------------------------------------------------------------
__global__ void __launch_bounds__(128)
conv3x3_wmma_k(const h16* __restrict__ x, const h16* __restrict__ w16,
               const float* __restrict__ bias, void* __restrict__ outp,
               int Cout, int Coutp, int H, int W, int Cinp, int mode)
{
    __shared__ __align__(64) h16 Bs[2][64][32];   // double-buffered B slab

    const int HW    = H * W;
    const int lane  = threadIdx.x & 31;
    const int wave  = threadIdx.x >> 5;
    const int pBase = blockIdx.x * 64;
    const int coBase = blockIdx.y * 64 + wave * 16;

    const int nC     = lane & 15;                 // C-layout column
    const int kBaseA = (lane < 16) ? 0 : 16;      // A-layout K base (contiguous, matches B)
    const int kOffB  = (lane < 16) ? 0 : 16;      // B-layout K base
    const int mHalf  = (lane >> 4) * 8;           // C-layout row offset

    // A row: lane supplies out-channel row (always < Coutp; pad rows are zero)
    const int co_a = coBase + (lane & 15);

    // B staging coords: thread t -> pixel n = t&63, k-range kk = (t>>6)*16
    const int sn = threadIdx.x & 63;
    const int kk = (threadIdx.x >> 6) * 16;
    const int sp = pBase + sn;
    const int spy = sp / W;
    const int spx = sp - spy * W;

    // tap geometry: validity + address-safe pixel offset
    auto geo = [&](int tap, bool& valid, long& qs) {
        int t3 = tap / 3;
        int dy = t3 - 1;
        int dx = (tap - t3 * 3) - 1;
        int yy = spy + dy, xx = spx + dx;
        valid = (sp < HW) && ((unsigned)yy < (unsigned)H) &&
                ((unsigned)xx < (unsigned)W);
        long q = (long)sp + (long)dy * W + dx;
        qs = valid ? q : 0;                        // always in-bounds
    };

    // stage 64px x 32ch slab into Bs[buf]; branchless, pointer-strided
    auto stage = [&](int buf, bool valid, long qs, int ci0) {
        const h16* bp = x + (size_t)((long)(ci0 + kk) * HW + qs);
        h16* dst = &Bs[buf][sn][kk];
        #pragma unroll
        for (int i = 0; i < 16; i += 2) {
            h16 va = bp[0];
            h16 vb = bp[HW];
            bp += 2 * (size_t)HW;
            h16x2 pk;
            pk[0] = valid ? va : (h16)0.0f;        // v_cndmask, no branch
            pk[1] = valid ? vb : (h16)0.0f;
            *(h16x2*)(dst + i) = pk;
        }
    };

    v8f acc0 = {}, acc1 = {}, acc2 = {}, acc3 = {};

    const int nci = Cinp >> 5;
    const int Nch = 9 * nci;

    bool stValid; long stQs;
    geo(0, stValid, stQs);
    stage(0, stValid, stQs, 0);
    __syncthreads();

    int ldTap = 0, ldCi = 0;     // chunk being consumed
    int stTap = 0, stCi = 0;     // chunk most recently staged

    for (int t = 0; t < Nch; ++t) {
        const int cur = t & 1;

        // ---- A slab in contiguous K order (K = 16h + s), same as B ----
        const h16* ap = w16 + ((size_t)ldTap * Coutp + co_a) * (size_t)Cinp
                        + (size_t)(ldCi << 5) + kBaseA;
        v8h alo = *(const v8h*)(ap);        // K = kb + 0..7
        v8h ahi = *(const v8h*)(ap + 8);    // K = kb + 8..15
        __builtin_prefetch(ap + 32, 0, 0);  // global_prefetch_b8
        v16h a;
        #pragma unroll
        for (int e = 0; e < 8; ++e) { a[e] = alo[e]; a[8 + e] = ahi[e]; }

        // ---- B regs from current LDS buffer ----
        v16h b0 = *(const v16h*)(&Bs[cur][nC +  0][kOffB]);
        v16h b1 = *(const v16h*)(&Bs[cur][nC + 16][kOffB]);
        v16h b2 = *(const v16h*)(&Bs[cur][nC + 32][kOffB]);
        v16h b3 = *(const v16h*)(&Bs[cur][nC + 48][kOffB]);

        // ---- stage next chunk into the other buffer (overlaps WMMA) ----
        if (t + 1 < Nch) {
            ++stCi;
            if (stCi == nci) { stCi = 0; ++stTap; geo(stTap, stValid, stQs); }
            stage(cur ^ 1, stValid, stQs, stCi << 5);
        }

        // ---- 4 WMMAs sharing A over the 64-pixel N tile ----
        acc0 = __builtin_amdgcn_wmma_f32_16x16x32_f16(false, a, false, b0,
                    (short)0, acc0, false, false);
        acc1 = __builtin_amdgcn_wmma_f32_16x16x32_f16(false, a, false, b1,
                    (short)0, acc1, false, false);
        acc2 = __builtin_amdgcn_wmma_f32_16x16x32_f16(false, a, false, b2,
                    (short)0, acc2, false, false);
        acc3 = __builtin_amdgcn_wmma_f32_16x16x32_f16(false, a, false, b3,
                    (short)0, acc3, false, false);

        ++ldCi;
        if (ldCi == nci) { ldCi = 0; ++ldTap; }
        __syncthreads();
    }

    // ---- epilogue: C layout lanes 0-15 -> M=r, lanes 16-31 -> M=8+r ----
    const v8f* accs[4] = { &acc0, &acc1, &acc2, &acc3 };
    #pragma unroll
    for (int g = 0; g < 4; ++g) {
        int p = pBase + g * 16 + nC;
        if (p >= HW) continue;
        #pragma unroll
        for (int r = 0; r < 8; ++r) {
            int co = coBase + mHalf + r;
            if (co < Cout) {
                float v = (*accs[g])[r] + bias[co];
                v = (v > 0.0f) ? v : 0.0f;
                if (mode == 0)
                    ((h16*)outp)[(size_t)co * HW + p] = (h16)v;
                else
                    ((float*)outp)[(size_t)co * HW + p] = tanhf(v);
            }
        }
    }
}

// ---------------------------------------------------------------------------
// 2x2 max pool, stride 2 (f16)
// ---------------------------------------------------------------------------
__global__ void maxpool2_k(const h16* __restrict__ x, h16* __restrict__ y,
                           int C, int H, int W) {
    int Ho = H >> 1, Wo = W >> 1;
    int total = C * Ho * Wo;
    int i = blockIdx.x * 256 + threadIdx.x;
    if (i >= total) return;
    int c = i / (Ho * Wo);
    int r = i - c * (Ho * Wo);
    int yo = r / Wo, xo = r - yo * Wo;
    const h16* p = x + (size_t)c * H * W + (size_t)(yo * 2) * W + xo * 2;
    float a = (float)p[0], b = (float)p[1], cc = (float)p[W], d = (float)p[W + 1];
    y[i] = (h16)fmaxf(fmaxf(a, b), fmaxf(cc, d));
}

// ---------------------------------------------------------------------------
// out = concat([skip (Cs @ 2h x 2w), bilinear_up2x(x (Cx @ h x w))], ch-axis)
// half-pixel convention (align_corners=False), edge clamp.
// ---------------------------------------------------------------------------
__global__ void upcat_k(const h16* __restrict__ skip, const h16* __restrict__ x,
                        h16* __restrict__ out, int Cs, int Cx, int h, int w) {
    int H = 2 * h, W = 2 * w;
    int total = (Cs + Cx) * H * W;
    int i = blockIdx.x * 256 + threadIdx.x;
    if (i >= total) return;
    int c = i / (H * W);
    int r = i - c * (H * W);
    if (c < Cs) { out[i] = skip[(size_t)c * H * W + r]; return; }
    int cx = c - Cs;
    int yo = r / W, xo = r - yo * W;
    float sy = (yo + 0.5f) * 0.5f - 0.5f;
    float sx = (xo + 0.5f) * 0.5f - 0.5f;
    int y0 = (int)floorf(sy), x0 = (int)floorf(sx);
    float fy = sy - y0, fx = sx - x0;
    int y1 = y0 + 1, x1 = x0 + 1;
    y0 = max(0, min(h - 1, y0)); y1 = max(0, min(h - 1, y1));
    x0 = max(0, min(w - 1, x0)); x1 = max(0, min(w - 1, x1));
    const h16* b = x + (size_t)cx * h * w;
    float v00 = (float)b[(size_t)y0 * w + x0], v01 = (float)b[(size_t)y0 * w + x1];
    float v10 = (float)b[(size_t)y1 * w + x0], v11 = (float)b[(size_t)y1 * w + x1];
    float v = v00 * (1.f - fy) * (1.f - fx) + v01 * (1.f - fy) * fx
            + v10 * fy * (1.f - fx)         + v11 * fy * fx;
    out[i] = (h16)v;
}

// ---------------------------------------------------------------------------
// Host orchestration
// ---------------------------------------------------------------------------
struct LayerCh { int ci, co; };
static const LayerCh CHS[31] = {
    {3, 32},
    {32, 32}, {32, 32},
    {32, 64}, {64, 64},
    {64, 64}, {64, 64},
    {64, 128}, {128, 128},
    {128, 128}, {128, 128},
    {128, 256},
    {256, 256},
    {384, 256}, {256, 256}, {256, 256},
    {384, 128}, {128, 128}, {128, 128},
    {192, 64}, {64, 64}, {64, 64},
    {128, 64}, {64, 64}, {64, 64},
    {96, 64}, {64, 64}, {64, 64},
    {64, 128},
    {128, 128},
    {128, 225},
};

extern "C" void kernel_launch(void* const* d_in, const int* in_sizes, int n_in,
                              void* d_out, int out_size, void* d_ws, size_t ws_size,
                              hipStream_t stream) {
    (void)in_sizes; (void)n_in; (void)out_size; (void)ws_size;

    char* ws = (char*)d_ws;
    size_t off = 0;
    auto alloc_h = [&](size_t nhalf) -> h16* {
        h16* p = (h16*)(ws + off);
        off += nhalf * sizeof(h16);
        off = (off + 255) & ~(size_t)255;
        return p;
    };

    // ---- weight repack (fp32 OIHW -> f16 [tap][Coutp][Cinp]) ----
    h16* wbuf[31];
    for (int i = 0; i < 31; ++i) {
        int Cin = CHS[i].ci, Cout = CHS[i].co;
        int Cinp  = ((Cin  + 31) / 32) * 32;
        int Coutp = ((Cout + 63) / 64) * 64;
        int total = 9 * Coutp * Cinp;
        wbuf[i] = alloc_h((size_t)total);
        wpack_k<<<(total + 255) / 256, 256, 0, stream>>>(
            (const float*)d_in[1 + 2 * i], wbuf[i], Cin, Cout, Cinp, Coutp, total);
    }

    // ---- activation buffers (channel counts all multiples of 32) ----
    const int HW0 = 320 * 320;                 // 102400
    h16* in16 = alloc_h((size_t)32 * HW0);           // 3 real + 29 zero pad
    h16* sd1  = alloc_h((size_t)32  * HW0);          // 320
    h16* sd2  = alloc_h((size_t)64  * 160 * 160);    // 160
    h16* sd3  = alloc_h((size_t)64  * 80  * 80);     // 80
    h16* sd4  = alloc_h((size_t)128 * 40  * 40);     // 40
    h16* sd5  = alloc_h((size_t)128 * 20  * 20);     // 20
    h16* pa   = alloc_h((size_t)128 * HW0);          // ping (max 26 MB)
    h16* pb   = alloc_h((size_t)128 * HW0);          // pong

    auto conv = [&](const h16* in, int li, void* out, int H, int W, int mode) {
        int Cin = CHS[li].ci, Cout = CHS[li].co;
        int Cinp  = ((Cin  + 31) / 32) * 32;
        int Coutp = ((Cout + 63) / 64) * 64;
        dim3 grid((H * W + 63) / 64, Coutp / 64);
        conv3x3_wmma_k<<<grid, 128, 0, stream>>>(
            in, wbuf[li], (const float*)d_in[2 + 2 * li], out,
            Cout, Coutp, H, W, Cinp, mode);
    };
    auto pool = [&](const h16* in, h16* out, int C, int H, int W) {
        int total = C * (H / 2) * (W / 2);
        maxpool2_k<<<(total + 255) / 256, 256, 0, stream>>>(in, out, C, H, W);
    };
    auto upcat = [&](const h16* skip, const h16* x, h16* out, int Cs, int Cx, int h, int w) {
        int total = (Cs + Cx) * (2 * h) * (2 * w);
        upcat_k<<<(total + 255) / 256, 256, 0, stream>>>(skip, x, out, Cs, Cx, h, w);
    };

    // input f32 -> f16 with zero-padded channels 3..31
    cvt_f32_f16_k<<<(32 * HW0 + 255) / 256, 256, 0, stream>>>(
        (const float*)d_in[0], in16, 3 * HW0, 32 * HW0);

    // ---- encoder ----
    conv(in16, 0, pa, 320, 320, 0);                  // x0: 32@320 in pa
    conv(pa, 1, pb, 320, 320, 0);
    conv(pb, 2, sd1, 320, 320, 0);                   // d1
    pool(sd1, pa, 32, 320, 320);
    conv(pa, 3, pb, 160, 160, 0);
    conv(pb, 4, sd2, 160, 160, 0);                   // d2
    pool(sd2, pa, 64, 160, 160);
    conv(pa, 5, pb, 80, 80, 0);
    conv(pb, 6, sd3, 80, 80, 0);                     // d3
    pool(sd3, pa, 64, 80, 80);
    conv(pa, 7, pb, 40, 40, 0);
    conv(pb, 8, sd4, 40, 40, 0);                     // d4
    pool(sd4, pa, 128, 40, 40);
    conv(pa, 9, pb, 20, 20, 0);
    conv(pb, 10, sd5, 20, 20, 0);                    // d5
    pool(sd5, pa, 128, 20, 20);
    conv(pa, 11, pb, 10, 10, 0);
    conv(pb, 12, pa, 10, 10, 0);                     // f: 256@10 in pa

    // ---- decoder ----
    upcat(sd5, pa, pb, 128, 256, 10, 10);            // 384@20
    conv(pb, 13, pa, 20, 20, 0);
    conv(pa, 14, pb, 20, 20, 0);
    conv(pb, 15, pa, 20, 20, 0);                     // u1: 256@20

    upcat(sd4, pa, pb, 128, 256, 20, 20);            // 384@40
    conv(pb, 16, pa, 40, 40, 0);
    conv(pa, 17, pb, 40, 40, 0);
    conv(pb, 18, pa, 40, 40, 0);                     // u2: 128@40

    upcat(sd3, pa, pb, 64, 128, 40, 40);             // 192@80
    conv(pb, 19, pa, 80, 80, 0);
    conv(pa, 20, pb, 80, 80, 0);
    conv(pb, 21, pa, 80, 80, 0);                     // u3: 64@80

    upcat(sd2, pa, pb, 64, 64, 80, 80);              // 128@160
    conv(pb, 22, pa, 160, 160, 0);
    conv(pa, 23, pb, 160, 160, 0);
    conv(pb, 24, pa, 160, 160, 0);                   // u4: 64@160

    upcat(sd1, pa, pb, 32, 64, 160, 160);            // 96@320
    conv(pb, 25, pa, 320, 320, 0);
    conv(pa, 26, pb, 320, 320, 0);
    conv(pb, 27, pa, 320, 320, 0);                   // u5: 64@320

    conv(pa, 28, pb, 320, 320, 0);                   // 128@320
    conv(pb, 29, pa, 320, 320, 0);                   // 128@320
    conv(pa, 30, d_out, 320, 320, 1);                // 225@320, fp32 tanh(relu)
}
